// GraphSAGE_52931176956147
// MI455X (gfx1250) — matrix operand order repaired and checked by reading, not compiled, not run on hardware
//
#include <hip/hip_runtime.h>
#include <hip/hip_bf16.h>

#define NNODES 100000
#define NEDGES 1600000

typedef __attribute__((ext_vector_type(16))) _Float16 v16h;
typedef __attribute__((ext_vector_type(4)))  _Float16 v4h;
typedef __attribute__((ext_vector_type(8)))  float    v8f;

// ---------------------------------------------------------------------------
// A fragment: 16x32 f16 tile stored row-major [16][ldk] halfs in LDS.
// Lane layout: lane L holds row M=L%16; K
//   {0..15} for lanes 0..15, {16..31} for lanes 16..31.
// One contiguous 16-half chunk.
// ---------------------------------------------------------------------------
__device__ __forceinline__ v16h load_frag_a(const _Float16* tile, int ldk,
                                            int k0, int lane) {
  const int hsel = (lane >> 4) & 1;
  const int m    = lane & 15;
  const _Float16* r = tile + m * ldk + k0 + hsel * 16;
  v16h a;
#pragma unroll
  for (int v = 0; v < 8; ++v) {
    const int kb = 2 * v;
    a[2 * v]     = r[kb];
    a[2 * v + 1] = r[kb + 1];
  }
  return a;
}

// ---------------------------------------------------------------------------
// Degree: one thread per edge
// ---------------------------------------------------------------------------
__global__ void __launch_bounds__(256)
deg_kernel(const long long* __restrict__ dst, float* __restrict__ deg, int nE) {
  int e = blockIdx.x * blockDim.x + threadIdx.x;
  if (e < nE) atomicAdd(&deg[(int)dst[e]], 1.0f);
}

// ---------------------------------------------------------------------------
// Scatter-add of gathered source rows: one thread per (edge, float4 chunk)
// ---------------------------------------------------------------------------
__global__ void __launch_bounds__(256)
scatter_kernel(const float* __restrict__ feat, const long long* __restrict__ src,
               const long long* __restrict__ dst, float* __restrict__ ns, int nE) {
  long long tid = (long long)blockIdx.x * blockDim.x + threadIdx.x;
  if (tid >= (long long)nE * 16) return;
  const int e = (int)(tid >> 4);
  const int c = ((int)tid & 15) << 2;
  const long long s = src[e];
  const long long d = dst[e];
  __builtin_prefetch(&src[e + 4096], 0, 0);   // global_prefetch_b8
  const float4 v = *(const float4*)(feat + s * 64 + c);
  float* o = ns + d * 64 + c;
  atomicAdd(o + 0, v.x);
  atomicAdd(o + 1, v.y);
  atomicAdd(o + 2, v.z);
  atomicAdd(o + 3, v.w);
}

// ---------------------------------------------------------------------------
// Fused SAGE layer: H = relu([X ; NS/deg] @ [Ws;Wn] + (bs+bn))
// 256 threads = 8 wave32; wave w owns 16 nodes x 64 cols.
// K = 128, tiled 4(nt) x 4(kt) x wmma_f32_16x16x32_f16.
// ---------------------------------------------------------------------------
__global__ void __launch_bounds__(256)
sage_combine(const float* __restrict__ X, const float* __restrict__ NS,
             const float* __restrict__ deg,
             const float* __restrict__ Ws, const float* __restrict__ bs,
             const float* __restrict__ Wn, const float* __restrict__ bn,
             float* __restrict__ H, int nNodes) {
  __shared__ _Float16 bfrag[4 * 4 * 32 * 16]; // [nt][kt][lane][i] 16KB
  __shared__ float    bsum[64];
  __shared__ _Float16 atile[8][16 * 128];     // per-wave A tiles, 32KB

  const int tid  = threadIdx.x;
  const int lane = tid & 31;
  const int wave = tid >> 5;

  // Stage weights directly in WMMA-B per-lane layout:
  //   n = nt*16 + lane%16 ; k = kt*32 + (lane/16)*16 + i ; k<64 -> Ws else Wn
  for (int idx = tid; idx < 4 * 4 * 32 * 16; idx += 256) {
    const int i  = idx & 15;
    const int ln = (idx >> 4) & 31;
    const int kt = (idx >> 9) & 3;
    const int nt = idx >> 11;
    const int n  = nt * 16 + (ln & 15);
    const int k  = kt * 32 + ((ln >> 4) & 1) * 16 + i;
    const float wv = (k < 64) ? Ws[k * 64 + n] : Wn[(k - 64) * 64 + n];
    bfrag[idx] = (_Float16)wv;
  }
  for (int i = tid; i < 64; i += 256) bsum[i] = bs[i] + bn[i];

  const int  nodeBase = blockIdx.x * 128 + wave * 16;
  const bool fullTile = (nodeBase + 16 <= nNodes);
  _Float16* at = atile[wave];

  // Build this wave's A tile: [16 nodes][128 = x(64) | neigh_mean(64)]
  for (int idx = lane; idx < 16 * 32; idx += 32) {
    const int r  = idx >> 5;
    const int c4 = (idx & 31) << 2;
    const int node = nodeBase + r;
    float4 v = make_float4(0.f, 0.f, 0.f, 0.f);
    if (fullTile || node < nNodes) {
      if (c4 < 64) {
        v = *(const float4*)(X + node * 64 + c4);
      } else {
        const float d   = deg[node];
        const float inv = 1.0f / (d > 1.0f ? d : 1.0f);
        v = *(const float4*)(NS + node * 64 + (c4 - 64));
        v.x *= inv; v.y *= inv; v.z *= inv; v.w *= inv;
      }
    }
    v4h h;
    h[0] = (_Float16)v.x; h[1] = (_Float16)v.y;
    h[2] = (_Float16)v.z; h[3] = (_Float16)v.w;
    *(v4h*)(at + r * 128 + c4) = h;
  }
  __syncthreads();

  // Preload all 4 A fragments once (reused across the 4 column tiles).
  v16h afr[4];
#pragma unroll
  for (int kt = 0; kt < 4; ++kt) afr[kt] = load_frag_a(at, 128, kt * 32, lane);

  const int rbase = (lane >> 4) * 8;
#pragma unroll
  for (int nt = 0; nt < 4; ++nt) {
    const float bias = bsum[nt * 16 + (lane & 15)];
    v8f acc;
#pragma unroll
    for (int v = 0; v < 8; ++v) acc[v] = bias;
#pragma unroll
    for (int kt = 0; kt < 4; ++kt) {
      const v16h bf =
          *(const v16h*)(bfrag + ((((nt << 2) | kt) << 5) + lane) * 16);
      acc = __builtin_amdgcn_wmma_f32_16x16x32_f16(
          false, afr[kt], false, bf, (short)0, acc, false, false);
    }
    const int col = nt * 16 + (lane & 15);
    float* hp = H + (nodeBase + rbase) * 64 + col;
    if (fullTile) {
      // Uniform fast path: no per-element exec manipulation.
#pragma unroll
      for (int v = 0; v < 8; ++v) hp[v * 64] = fmaxf(acc[v], 0.0f);
    } else {
#pragma unroll
      for (int v = 0; v < 8; ++v) {
        if (nodeBase + rbase + v < nNodes) hp[v * 64] = fmaxf(acc[v], 0.0f);
      }
    }
  }
}

// ---------------------------------------------------------------------------
// Output head: Y = H @ Wo + bo    (K=64, OUT=32, no relu)
// ---------------------------------------------------------------------------
__global__ void __launch_bounds__(256)
sage_out(const float* __restrict__ H, const float* __restrict__ Wo,
         const float* __restrict__ bo, float* __restrict__ Y, int nNodes) {
  __shared__ _Float16 bfrag[2 * 2 * 32 * 16]; // [nt][kt][lane][i] 4KB
  __shared__ float    bsh[32];
  __shared__ _Float16 atile[8][16 * 64];      // 16KB

  const int tid  = threadIdx.x;
  const int lane = tid & 31;
  const int wave = tid >> 5;

  for (int idx = tid; idx < 2 * 2 * 32 * 16; idx += 256) {
    const int i  = idx & 15;
    const int ln = (idx >> 4) & 31;
    const int kt = (idx >> 9) & 1;
    const int nt = idx >> 10;
    const int n  = nt * 16 + (ln & 15);
    const int k  = kt * 32 + ((ln >> 4) & 1) * 16 + i;
    bfrag[idx] = (_Float16)Wo[k * 32 + n];
  }
  if (tid < 32) bsh[tid] = bo[tid];

  const int  nodeBase = blockIdx.x * 128 + wave * 16;
  const bool fullTile = (nodeBase + 16 <= nNodes);
  _Float16* at = atile[wave];

  for (int idx = lane; idx < 16 * 16; idx += 32) {
    const int r  = idx >> 4;
    const int c4 = (idx & 15) << 2;
    const int node = nodeBase + r;
    float4 v = make_float4(0.f, 0.f, 0.f, 0.f);
    if (fullTile || node < nNodes) v = *(const float4*)(H + node * 64 + c4);
    v4h h;
    h[0] = (_Float16)v.x; h[1] = (_Float16)v.y;
    h[2] = (_Float16)v.z; h[3] = (_Float16)v.w;
    *(v4h*)(at + r * 64 + c4) = h;
  }
  __syncthreads();

  v16h afr[2];
#pragma unroll
  for (int kt = 0; kt < 2; ++kt) afr[kt] = load_frag_a(at, 64, kt * 32, lane);

  const int rbase = (lane >> 4) * 8;
#pragma unroll
  for (int nt = 0; nt < 2; ++nt) {
    const float bias = bsh[nt * 16 + (lane & 15)];
    v8f acc;
#pragma unroll
    for (int v = 0; v < 8; ++v) acc[v] = bias;
#pragma unroll
    for (int kt = 0; kt < 2; ++kt) {
      const v16h bf =
          *(const v16h*)(bfrag + ((((nt << 1) | kt) << 5) + lane) * 16);
      acc = __builtin_amdgcn_wmma_f32_16x16x32_f16(
          false, afr[kt], false, bf, (short)0, acc, false, false);
    }
    const int col = nt * 16 + (lane & 15);
    float* yp = Y + (nodeBase + rbase) * 32 + col;
    if (fullTile) {
#pragma unroll
      for (int v = 0; v < 8; ++v) yp[v * 32] = acc[v];
    } else {
#pragma unroll
      for (int v = 0; v < 8; ++v) {
        if (nodeBase + rbase + v < nNodes) yp[v * 32] = acc[v];
      }
    }
  }
}

// ---------------------------------------------------------------------------
extern "C" void kernel_launch(void* const* d_in, const int* in_sizes, int n_in,
                              void* d_out, int out_size, void* d_ws, size_t ws_size,
                              hipStream_t stream) {
  const float*     x   = (const float*)d_in[0];
  const long long* ei  = (const long long*)d_in[1];   // int64 [2, NEDGES]
  const float*     Ws1 = (const float*)d_in[2];
  const float*     bs1 = (const float*)d_in[3];
  const float*     Wn1 = (const float*)d_in[4];
  const float*     bn1 = (const float*)d_in[5];
  const float*     Ws2 = (const float*)d_in[6];
  const float*     bs2 = (const float*)d_in[7];
  const float*     Wn2 = (const float*)d_in[8];
  const float*     bn2 = (const float*)d_in[9];
  const float*     Wo  = (const float*)d_in[10];
  const float*     bo  = (const float*)d_in[11];

  const long long* src = ei;
  const long long* dst = ei + NEDGES;

  auto align256 = [](size_t v) { return (v + 255) & ~(size_t)255; };
  char*  ws  = (char*)d_ws;
  size_t off = 0;
  float* deg = (float*)(ws + off); off += align256(sizeof(float) * NNODES);
  float* ns  = (float*)(ws + off); off += align256(sizeof(float) * (size_t)NNODES * 64);
  float* h1  = (float*)(ws + off); off += align256(sizeof(float) * (size_t)NNODES * 64);
  float* h2  = (float*)(ws + off); off += align256(sizeof(float) * (size_t)NNODES * 64);
  (void)ws_size; (void)in_sizes; (void)n_in; (void)out_size;

  const int thr = 256;
  const long long scatterThreads = (long long)NEDGES * 16;
  const int scatterBlocks = (int)((scatterThreads + thr - 1) / thr);
  const int edgeBlocks    = (NEDGES + thr - 1) / thr;
  const int nodeBlocks    = (NNODES + 127) / 128;

  hipMemsetAsync(deg, 0, sizeof(float) * NNODES, stream);
  hipMemsetAsync(ns,  0, sizeof(float) * (size_t)NNODES * 64, stream);

  deg_kernel<<<edgeBlocks, thr, 0, stream>>>(dst, deg, NEDGES);

  // ---- Layer 1 ----
  scatter_kernel<<<scatterBlocks, thr, 0, stream>>>(x, src, dst, ns, NEDGES);
  sage_combine<<<nodeBlocks, thr, 0, stream>>>(x, ns, deg, Ws1, bs1, Wn1, bn1,
                                               h1, NNODES);

  // ---- Layer 2 ----
  hipMemsetAsync(ns, 0, sizeof(float) * (size_t)NNODES * 64, stream);
  scatter_kernel<<<scatterBlocks, thr, 0, stream>>>(h1, src, dst, ns, NEDGES);
  sage_combine<<<nodeBlocks, thr, 0, stream>>>(h1, ns, deg, Ws2, bs2, Wn2, bn2,
                                               h2, NNODES);

  // ---- Output head ----
  sage_out<<<nodeBlocks, thr, 0, stream>>>(h2, Wo, bo, (float*)d_out, NNODES);
}
